// ObjectDetectionHead_45861660787379
// MI455X (gfx1250) — matrix operand order repaired and checked by reading, not compiled, not run on hardware
//
#include <hip/hip_runtime.h>
#include <hip/hip_bf16.h>

typedef __attribute__((ext_vector_type(16))) _Float16 v16h;
typedef __attribute__((ext_vector_type(8)))  _Float16 v8h;
typedef __attribute__((ext_vector_type(8)))  float    v8f;

#define S 64
#define SS (64*64)
#define VOL (64*64*64)
#define CIN 64
#define NCLS 5
#define NOFF 3
#define KTOT 1728            // 27 taps * 64 cin
#define NGT 64
#define STRIDE_F 32.0f
#define LOGITS_ELEMS (2*NCLS*VOL)
#define OFF_ELEMS    (2*NOFF*VOL)
#define LOSS_IDX (LOGITS_ELEMS + OFF_ELEMS)

#define XH_ELEMS (3*4*66*64)   // halo halves: zz(3) * yy(4) * x'(66) * cin(64)

// ---------------- prep: reorder weights (5+3)x64x3x3x3 fp32 -> 16 x 1728 f16 (rows 8..15 zero)
__global__ void prep_weights_k(const float* __restrict__ cls_w,
                               const float* __restrict__ off_w,
                               _Float16* __restrict__ wpack) {
  int i = blockIdx.x * 256 + threadIdx.x;
  if (i >= 16 * KTOT) return;
  int m = i / KTOT;
  int k = i - m * KTOT;
  int tap = k >> 6;          // kz*9+ky*3+kx
  int cin = k & 63;
  float v = 0.0f;
  if (m < NCLS)      v = cls_w[(m * CIN + cin) * 27 + tap];
  else if (m < 8)    v = off_w[((m - NCLS) * CIN + cin) * 27 + tap];
  wpack[i] = (_Float16)v;
}

// ---------------- conv as implicit GEMM via v_wmma_f32_16x16x32_f16
// grid: B*S*(S/2) blocks: one (b,z) and TWO y-rows per block.
// 128 threads = 4 waves; wave w owns x-tile [16w,16w+16) for BOTH rows (2 accumulators,
// sharing each A fragment between the two WMMAs).
__global__ __launch_bounds__(128) void conv_wmma_k(
    const float* __restrict__ feat, const _Float16* __restrict__ wpack,
    const float* __restrict__ cls_b, const float* __restrict__ off_b,
    float* __restrict__ out)
{
  extern __shared__ _Float16 smem[];
  _Float16* wlds = smem;                 // 16*1728 halves
  _Float16* xlds = smem + 16 * KTOT;     // 3*4*66*64 halves, cin contiguous

  const int tid = threadIdx.x;
  const int blk = blockIdx.x;
  const int b = blk >> 11;               // 2048 blocks per batch
  const int rem = blk & 2047;
  const int z = rem >> 5;
  const int y0 = (rem & 31) << 1;        // two rows: y0, y0+1

  const float* fb = feat + (size_t)b * CIN * VOL;

  // warm the halo cache lines (global_prefetch_b8): 64c * 12(zz,yy) rows * 2 lines
  #pragma unroll 1
  for (int i = tid; i < 64 * 12 * 2; i += 128) {
    int l = i & 1;
    int t = i >> 1;
    int r = t % 12;
    int c = t / 12;
    int gz = z + (r >> 2) - 1;
    int gy = y0 + (r & 3) - 1;
    if ((unsigned)gz < 64u && (unsigned)gy < 64u)
      __builtin_prefetch(fb + (((c * S + gz) * S + gy) * S) + l * 32, 0, 3);
  }

  // stage weights (f16, already packed): 55296 B via uint4
  {
    const uint4* src = (const uint4*)wpack;
    uint4* dst = (uint4*)wlds;
    #pragma unroll 1
    for (int i = tid; i < (16 * KTOT) / 8; i += 128) dst[i] = src[i];
  }
  // stage input halo: zz(3) yy(4) x'(66) cin(64), fp32 -> f16, zero pad at borders
  #pragma unroll 1
  for (int idx = tid; idx < XH_ELEMS; idx += 128) {
    int x = idx % 66;
    int r = idx / 66;
    int c = r & 63;
    int zzyy = r >> 6;                   // 0..11
    int zz = zzyy >> 2;
    int yy = zzyy & 3;
    int gz = z + zz - 1, gy = y0 + yy - 1, gx = x - 1;
    float v = 0.0f;
    if ((unsigned)gz < 64u && (unsigned)gy < 64u && (unsigned)gx < 64u)
      v = fb[((c * S + gz) * S + gy) * S + gx];
    xlds[(zzyy * 66 + x) * 64 + c] = (_Float16)v;
  }
  __syncthreads();

  const int lane = tid & 31;
  const int wave = tid >> 5;
  const int col  = lane & 15;            // A row (channel) and B column (voxel)
  const int khi  = lane >> 4;            // lane-group K split per ISA layout
  const int x0   = wave << 4;

  const _Float16* wrow = wlds + col * KTOT;
  const int x0col = x0 + col;

  v8f acc[2];
  #pragma unroll
  for (int t = 0; t < 2; ++t)
    #pragma unroll
    for (int r = 0; r < 8; ++r) acc[t][r] = 0.0f;

  #pragma unroll 1
  for (int tap = 0; tap < 27; ++tap) {
    int kz = tap / 9;
    int kyx = tap - kz * 9;
    int ky = kyx / 3;
    int kx = kyx - ky * 3;
    #pragma unroll
    for (int h = 0; h < 2; ++h) {
      int kb = tap * 64 + h * 32;
      // A fragment: lanes<16 hold K {0..7, 16..23}, lanes>=16 hold K {8..15, 24..31}
      v8h alo = *(const v8h*)(wrow + kb + khi * 8);
      v8h ahi = *(const v8h*)(wrow + kb + 16 + khi * 8);
      v16h a = __builtin_shufflevector(alo, ahi, 0,1,2,3,4,5,6,7,8,9,10,11,12,13,14,15);
      #pragma unroll
      for (int t = 0; t < 2; ++t) {
        // B fragment: lane holds column N=col, K (= cin) {0..7, 16..23} (lanes<16) / {8..15, 24..31} (lanes>=16)
        int vox = ((kz * 4 + t + ky) * 66 + x0col + kx) * 64;
        v8h blo = *(const v8h*)(xlds + vox + h * 32 + khi * 8);
        v8h bhi = *(const v8h*)(xlds + vox + h * 32 + 16 + khi * 8);
        v16h bm = __builtin_shufflevector(blo, bhi, 0,1,2,3,4,5,6,7,8,9,10,11,12,13,14,15);
        acc[t] = __builtin_amdgcn_wmma_f32_16x16x32_f16(false, a, false, bm,
                                                        (short)0, acc[t], false, false);
      }
    }
  }

  // D layout: lane = N (voxel), VGPR r = channel r (lanes<16) / r+8 (lanes>=16, all padding)
  if (lane < 16) {
    int x = x0 + col;
    float* logits = out;
    float* offs   = out + LOGITS_ELEMS;
    #pragma unroll
    for (int t = 0; t < 2; ++t) {
      size_t vox = (size_t)(z * SS + (y0 + t) * S + x);
      #pragma unroll
      for (int r = 0; r < 8; ++r) {
        float v = acc[t][r];
        if (r < NCLS) logits[((size_t)(b * NCLS + r)) * VOL + vox] = v + cls_b[r];
        else          offs[((size_t)(b * NOFF + (r - NCLS))) * VOL + vox] = v + off_b[r - NCLS];
      }
    }
  }
}

// ---------------- per-GT nearest-anchor argmin (argmax of iou == argmin of d)
__global__ __launch_bounds__(256) void match_k(
    const float* __restrict__ labels, const float* __restrict__ out,
    int* __restrict__ bestA)
{
  int b = blockIdx.x >> 6;
  int n = blockIdx.x & 63;
  const float* lb = labels + b * 5 * NGT;
  float tz = lb[0 * NGT + n], ty = lb[1 * NGT + n], tx = lb[2 * NGT + n];
  const float* off = out + LOGITS_ELEMS + (size_t)b * NOFF * VOL;
  float bd = 3.4e38f;
  int   bm = 0;
  for (int m = threadIdx.x; m < VOL; m += 256) {
    int z = m >> 12, y = (m >> 6) & 63, x = m & 63;
    float pz = ((float)z + 0.5f) * STRIDE_F + off[0 * VOL + m];
    float py = ((float)y + 0.5f) * STRIDE_F + off[1 * VOL + m];
    float px = ((float)x + 0.5f) * STRIDE_F + off[2 * VOL + m];
    float dz = pz - tz, dy = py - ty, dx = px - tx;
    float d = dz * dz + dy * dy + dx * dx;
    if (d < bd || (d == bd && m < bm)) { bd = d; bm = m; }
  }
  __shared__ float sd[256];
  __shared__ int   sm[256];
  sd[threadIdx.x] = bd; sm[threadIdx.x] = bm;
  __syncthreads();
  for (int s = 128; s > 0; s >>= 1) {
    if (threadIdx.x < s) {
      float od = sd[threadIdx.x + s]; int om = sm[threadIdx.x + s];
      if (od < sd[threadIdx.x] || (od == sd[threadIdx.x] && om < sm[threadIdx.x])) {
        sd[threadIdx.x] = od; sm[threadIdx.x] = om;
      }
    }
    __syncthreads();
  }
  if (threadIdx.x == 0) bestA[b * NGT + n] = sm[0];
}

__global__ void zero_loss_k(float* out) { out[LOSS_IDX] = 0.0f; }

// ---------------- focal + reg loss, block reduce, atomicAdd into loss slot
__global__ __launch_bounds__(256) void loss_k(
    const float* __restrict__ labels, const int* __restrict__ bestA,
    const int* __restrict__ num_items, float* __restrict__ out)
{
  __shared__ float stz[NGT], sty[NGT], stx[NGT], slb[NGT], ssg[NGT];
  __shared__ int   sba[NGT];
  int g = blockIdx.x * 256 + threadIdx.x;
  int b = g >> 18;
  int m = g & (VOL - 1);
  if (threadIdx.x < NGT) {
    const float* lb = labels + b * 5 * NGT;
    int n = threadIdx.x;
    stz[n] = lb[n];
    sty[n] = lb[NGT + n];
    stx[n] = lb[2 * NGT + n];
    slb[n] = lb[3 * NGT + n];
    ssg[n] = lb[4 * NGT + n];
    sba[n] = bestA[b * NGT + n];
  }
  __syncthreads();

  int matched = -1;
  #pragma unroll 1
  for (int n = 0; n < NGT; ++n)
    if (slb[n] != -100.0f && sba[n] == m) matched = n;   // ascending -> keeps max n
  int assigned = (matched >= 0) ? (int)slb[matched] : -1;

  const float* logits = out + (size_t)b * NCLS * VOL;
  float fsum = 0.0f;
  #pragma unroll
  for (int c = 0; c < NCLS; ++c) {
    float pl = logits[(size_t)c * VOL + m];
    float lse = log1pf(expf(-fabsf(pl)));
    float logp  = fminf(pl, 0.0f) - lse;      // log sigmoid(pl)
    float lognp = fminf(-pl, 0.0f) - lse;     // log sigmoid(-pl)
    float p = expf(logp);
    float t = (c == assigned) ? 1.0f : 0.0f;
    float q = 1.0f - p;
    fsum -= t * q * q * logp + (1.0f - t) * p * p * lognp;
  }

  float reg = 0.0f;
  if (matched < 0) {
    const float* off = out + LOGITS_ELEMS + (size_t)b * NOFF * VOL;
    int z = m >> 12, y = (m >> 6) & 63, x = m & 63;
    float pz = ((float)z + 0.5f) * STRIDE_F + off[0 * VOL + m];
    float py = ((float)y + 0.5f) * STRIDE_F + off[1 * VOL + m];
    float px = ((float)x + 0.5f) * STRIDE_F + off[2 * VOL + m];
    float dz = pz - stz[0], dy = py - sty[0], dx = px - stx[0];
    float d = dz * dz + dy * dy + dx * dx;
    float sg = ssg[0];
    reg = 1.0f - expf(-d / (2.0f * sg * sg));
  }

  __shared__ float red[256];
  red[threadIdx.x] = fsum + reg;
  __syncthreads();
  for (int s = 128; s > 0; s >>= 1) {
    if (threadIdx.x < s) red[threadIdx.x] += red[threadIdx.x + s];
    __syncthreads();
  }
  if (threadIdx.x == 0)
    atomicAdd(out + LOSS_IDX, red[0] / (float)num_items[0]);
}

extern "C" void kernel_launch(void* const* d_in, const int* in_sizes, int n_in,
                              void* d_out, int out_size, void* d_ws, size_t ws_size,
                              hipStream_t stream) {
  const float* feat   = (const float*)d_in[0];   // (2,64,64,64,64)
  const float* labels = (const float*)d_in[1];   // (2,5,64)
  const float* cls_w  = (const float*)d_in[2];   // (5,64,3,3,3)
  const float* cls_b  = (const float*)d_in[3];   // (5,)
  const float* off_w  = (const float*)d_in[4];   // (3,64,3,3,3)
  const float* off_b  = (const float*)d_in[5];   // (3,)
  const int*   n_items= (const int*)d_in[6];     // scalar 2
  float* out = (float*)d_out;

  _Float16* wpack = (_Float16*)d_ws;                    // 55296 B
  int* bestA = (int*)((char*)d_ws + 57344);             // 128 ints

  prep_weights_k<<<(16 * KTOT + 255) / 256, 256, 0, stream>>>(cls_w, off_w, wpack);

  size_t lds_bytes = (size_t)(16 * KTOT + XH_ELEMS) * sizeof(_Float16); // 156672 B
  conv_wmma_k<<<2 * S * (S / 2), 128, lds_bytes, stream>>>(feat, wpack, cls_b, off_b, out);

  match_k<<<2 * NGT, 256, 0, stream>>>(labels, out, bestA);
  zero_loss_k<<<1, 1, 0, stream>>>(out);
  loss_k<<<(2 * VOL) / 256, 256, 0, stream>>>(labels, bestA, n_items, out);
}
